// GatedGCNLayerSF_89129161327106
// MI455X (gfx1250) — compile-verified, matrix operand order checked by reading, not run on hardware
//
#include <hip/hip_runtime.h>
#include <math.h>

#define NN 10000
#define NE 320000
#define DD 128
#define FSIGMA 0.3f

typedef __attribute__((ext_vector_type(16))) _Float16 v16h;
typedef __attribute__((ext_vector_type(8)))  _Float16 v8h;
typedef __attribute__((ext_vector_type(8)))  float    v8f;

__device__ __forceinline__ v8f wmma16(v16h a, v16h b, v8f c) {
  // D = A(16x32 f16) * B(32x16 f16) + C(16x16 f32)
  return __builtin_amdgcn_wmma_f32_16x16x32_f16(false, a, false, b, (short)0, c,
                                                false, false);
}

// A fragment from an f32 row (row-major, K contiguous). Per CDNA5 ISA 16-bit A
// layout: lane holds M = lane&15; halves 0..7 = K = k0+half*8+{0..7},
// halves 8..15 = K = k0+16+half*8+{0..7}.
__device__ __forceinline__ v16h frag_a_f32(const float* __restrict__ row, int k0, int half) {
  const float4* p4 = (const float4*)(row + k0 + half * 8);
  float4 f0 = p4[0], f1 = p4[1], f2 = p4[4], f3 = p4[5];
  v16h a;
  a[0]=(_Float16)f0.x;  a[1]=(_Float16)f0.y;  a[2]=(_Float16)f0.z;  a[3]=(_Float16)f0.w;
  a[4]=(_Float16)f1.x;  a[5]=(_Float16)f1.y;  a[6]=(_Float16)f1.z;  a[7]=(_Float16)f1.w;
  a[8]=(_Float16)f2.x;  a[9]=(_Float16)f2.y;  a[10]=(_Float16)f2.z; a[11]=(_Float16)f2.w;
  a[12]=(_Float16)f3.x; a[13]=(_Float16)f3.y; a[14]=(_Float16)f3.z; a[15]=(_Float16)f3.w;
  return a;
}

// A fragment = elementwise product of two f32 rows (for h[src]*h[dst]).
__device__ __forceinline__ v16h frag_a_mul(const float* __restrict__ ra,
                                           const float* __restrict__ rb, int k0, int half) {
  const float4* pa = (const float4*)(ra + k0 + half * 8);
  const float4* pb = (const float4*)(rb + k0 + half * 8);
  float4 a0 = pa[0], a1 = pa[1], a2 = pa[4], a3 = pa[5];
  float4 b0 = pb[0], b1 = pb[1], b2 = pb[4], b3 = pb[5];
  v16h a;
  a[0]=(_Float16)(a0.x*b0.x);  a[1]=(_Float16)(a0.y*b0.y);
  a[2]=(_Float16)(a0.z*b0.z);  a[3]=(_Float16)(a0.w*b0.w);
  a[4]=(_Float16)(a1.x*b1.x);  a[5]=(_Float16)(a1.y*b1.y);
  a[6]=(_Float16)(a1.z*b1.z);  a[7]=(_Float16)(a1.w*b1.w);
  a[8]=(_Float16)(a2.x*b2.x);  a[9]=(_Float16)(a2.y*b2.y);
  a[10]=(_Float16)(a2.z*b2.z); a[11]=(_Float16)(a2.w*b2.w);
  a[12]=(_Float16)(a3.x*b3.x); a[13]=(_Float16)(a3.y*b3.y);
  a[14]=(_Float16)(a3.z*b3.z); a[15]=(_Float16)(a3.w*b3.w);
  return a;
}

// B fragment from f16 weight row [n][k] (lane holds N = lane&15, same K split).
__device__ __forceinline__ v16h frag_b16(const _Float16* __restrict__ wrow, int k0, int half) {
  const v8h* p = (const v8h*)(wrow + k0 + half * 8);
  v8h b0 = p[0], b1 = p[2];
  v16h b;
#pragma unroll
  for (int j = 0; j < 8; ++j) { b[j] = b0[j]; b[8 + j] = b1[j]; }
  return b;
}

__device__ __forceinline__ float lrelu(float x) { return x > 0.f ? x : 0.01f * x; }

// ---------------------------------------------------------------- utilities
__global__ void cvt_w_kernel(const float* __restrict__ src, _Float16* __restrict__ dst, int n) {
  int i = blockIdx.x * blockDim.x + threadIdx.x;
  if (i < n) dst[i] = (_Float16)src[i];
}

__global__ void zero_kernel(float* __restrict__ p, int n) {
  int i = blockIdx.x * blockDim.x + threadIdx.x;
  if (i < n) p[i] = 0.f;
}

// Histogram: count non-spatial edges per destination node.
__global__ void dst_count_kernel(const int* __restrict__ dst, const int* __restrict__ smask,
                                 float* __restrict__ cnt) {
  int i = blockIdx.x * blockDim.x + threadIdx.x;
  if (i < NE && smask[i] == 0) atomicAdd(&cnt[dst[i]], 1.0f);
}

// -------------------------------------------------- fused node GEMMs (WMMA)
// g = 0..6 : V1h,V2h,P1h,P2h,P3h,D1h,D2h  (X @ W^T + b), X in {h,p,d}
// g = 7    : tau_node = relu(concat(h,d) @ W_T^T + b_T)   (K = 256)
// Each wave: one 16-row tile, two 16-col subtiles, A fragments cached in VGPRs.
__global__ void node_gemm_kernel(const float* __restrict__ h, const float* __restrict__ p,
                                 const float* __restrict__ d,
                                 const _Float16* __restrict__ w16,
                                 const float* __restrict__ bV1, const float* __restrict__ bV2,
                                 const float* __restrict__ bP1, const float* __restrict__ bP2,
                                 const float* __restrict__ bP3, const float* __restrict__ bD1,
                                 const float* __restrict__ bD2, const float* __restrict__ bT,
                                 float* __restrict__ outbase) {
  int g = blockIdx.z;
  int wave = threadIdx.x >> 5, lane = threadIdx.x & 31;
  int half = lane >> 4, nn = lane & 15;
  int m0 = blockIdx.x * 16;
  float* out = outbase + (size_t)g * NN * DD;

  if (g < 7) {
    const float* X = (g < 2) ? h : (g < 5) ? p : d;
    const float* bias;
    switch (g) {
      case 0: bias = bV1; break; case 1: bias = bV2; break;
      case 2: bias = bP1; break; case 3: bias = bP2; break;
      case 4: bias = bP3; break; case 5: bias = bD1; break;
      default: bias = bD2;
    }
    const float* arow = X + (size_t)(m0 + nn) * DD;
    v16h af[4];
#pragma unroll
    for (int kk = 0; kk < 4; ++kk) af[kk] = frag_a_f32(arow, kk * 32, half);
#pragma unroll
    for (int sub = 0; sub < 2; ++sub) {
      int n = wave * 32 + sub * 16 + nn;
      const _Float16* brow = w16 + (size_t)g * DD * DD + (size_t)n * DD;
      v8f acc = {};
#pragma unroll
      for (int kk = 0; kk < 4; ++kk)
        acc = wmma16(af[kk], frag_b16(brow, kk * 32, half), acc);
#pragma unroll
      for (int r = 0; r < 8; ++r) {
        int m = r + 8 * half;
        out[(size_t)(m0 + m) * DD + n] = acc[r] + bias[n];
      }
    }
  } else {
    const float* hrow = h + (size_t)(m0 + nn) * DD;
    const float* drow = d + (size_t)(m0 + nn) * DD;
    v16h af[8];
#pragma unroll
    for (int kk = 0; kk < 8; ++kk) {
      const float* srow = (kk < 4) ? hrow : drow;
      af[kk] = frag_a_f32(srow, (kk * 32) & (DD - 1), half);
    }
#pragma unroll
    for (int sub = 0; sub < 2; ++sub) {
      int n = wave * 32 + sub * 16 + nn;
      const _Float16* brow = w16 + (size_t)9 * DD * DD + (size_t)n * 2 * DD;
      v8f acc = {};
#pragma unroll
      for (int kk = 0; kk < 8; ++kk)
        acc = wmma16(af[kk], frag_b16(brow, kk * 32, half), acc);
#pragma unroll
      for (int r = 0; r < 8; ++r) {
        int m = r + 8 * half;
        float v = acc[r] + bT[n];
        out[(size_t)(m0 + m) * DD + n] = v > 0.f ? v : 0.f;  // relu for tau
      }
    }
  }
}

// ------------------------------------------------------- edge kernel (WMMA)
// Per block: 16 edges. E1e & Vj GEMMs on WMMA (A fragments cached across the
// two column subtiles), e_new, VJP coefficient, scatter grad into fab, e_out.
__global__ void edge_kernel(const float* __restrict__ h, const float* __restrict__ e,
                            const float* __restrict__ dv,
                            const int* __restrict__ src, const int* __restrict__ dst,
                            const int* __restrict__ smask,
                            const float* __restrict__ ws_nodes, float* __restrict__ fab,
                            const _Float16* __restrict__ wE1, const _Float16* __restrict__ wV,
                            const float* __restrict__ bE1, const float* __restrict__ bV,
                            float* __restrict__ e_out) {
  __shared__ float lds_en[16][DD];
  __shared__ float lds_vj[16][DD];
  __shared__ int s_src[16], s_dst[16], s_sm[16];
  __shared__ float s_coef[16];

  int tid = threadIdx.x;
  size_t e0 = (size_t)blockIdx.x * 16;
  if (tid < 16) {
    s_src[tid] = src[e0 + tid];
    s_dst[tid] = dst[e0 + tid];
    s_sm[tid] = smask[e0 + tid];
  }
  __syncthreads();

  int wave = tid >> 5, lane = tid & 31, half = lane >> 4, nn = lane & 15;
  const float* P1h = ws_nodes + (size_t)2 * NN * DD;
  const float* P2h = ws_nodes + (size_t)3 * NN * DD;

  const float* erow = e + (e0 + nn) * DD;
  const float* hsrow = h + (size_t)s_src[nn] * DD;
  const float* hdrow = h + (size_t)s_dst[nn] * DD;

  // Cache A fragments (all 4 K-steps) once; reuse for both column subtiles.
  v16h aE[4], aV[4];
#pragma unroll
  for (int kk = 0; kk < 4; ++kk) {
    aE[kk] = frag_a_f32(erow, kk * 32, half);
    aV[kk] = frag_a_mul(hsrow, hdrow, kk * 32, half);
  }

#pragma unroll
  for (int sub = 0; sub < 2; ++sub) {
    int n = wave * 32 + sub * 16 + nn;
    const _Float16* bEw = wE1 + (size_t)n * DD;
    const _Float16* bVw = wV + (size_t)n * DD;

    v8f accE = {}, accV = {};
#pragma unroll
    for (int kk = 0; kk < 4; ++kk) {
      accE = wmma16(aE[kk], frag_b16(bEw, kk * 32, half), accE);
      accV = wmma16(aV[kk], frag_b16(bVw, kk * 32, half), accV);
    }
#pragma unroll
    for (int r = 0; r < 8; ++r) {
      int m = r + 8 * half;
      float en = 0.5f * (P1h[(size_t)s_dst[m] * DD + n] -
                         P2h[(size_t)s_src[m] * DD + n] + accE[r] + bE1[n]);
      lds_en[m][n] = en;
      float vj = accV[r] + bV[n];
      vj = vj > 0.f ? vj : 0.f;
      lds_vj[m][n] = s_sm[m] ? vj : 0.f;  // Vj nonzero only on spatial edges
    }
  }
  __syncthreads();

  if (tid < 16) {  // per-edge scalar VJP coefficient
    float s2 = 0.f, sv = 0.f;
    for (int n = 0; n < DD; ++n) {
      float v = lds_en[tid][n];
      s2 += v * v;
      sv += lds_vj[tid][n];
    }
    float nrm = fmaxf(sqrtf(s2), 1e-9f);
    float r = 0.5f * sqrtf(nrm);
    // grad_rab = -S * exp(-r/sigma)/sigma * 0.25/sqrt(nrm) * rab/nrm
    s_coef[tid] = -sv * expf(-r / FSIGMA) * 0.25f / (FSIGMA * sqrtf(nrm) * nrm);
  }
  __syncthreads();

  {
    int m = tid >> 3;
    int nb = (tid & 7) * 16;
    float coef = s_coef[m];
    size_t db = (size_t)s_dst[m] * DD;
    size_t eb = (e0 + m) * DD;
#pragma unroll
    for (int j = 0; j < 16; ++j) {
      int n = nb + j;
      float en = lds_en[m][n];
      atomicAdd(&fab[db + n], coef * en);  // fab = segsum(grad_rab, dst)
      e_out[eb + n] = e[eb + n] + lrelu(en);
    }
  }
}

// ------------------------------------------------------------- final update
__global__ void final_kernel(const float* __restrict__ h, const float* __restrict__ p,
                             const float* __restrict__ dv, const float* __restrict__ dt,
                             const float* __restrict__ ws_nodes,
                             const float* __restrict__ fab, const float* __restrict__ cnt,
                             float* __restrict__ out_h, float* __restrict__ out_p,
                             float* __restrict__ out_d) {
  __shared__ float red[DD];
  __shared__ float s_norm;
  int i = blockIdx.x;
  int n = threadIdx.x;
  size_t idx = (size_t)i * DD + n;
  const float* V1h = ws_nodes;
  const float* V2h = ws_nodes + (size_t)1 * NN * DD;
  const float* P3h = ws_nodes + (size_t)4 * NN * DD;
  const float* D1h = ws_nodes + (size_t)5 * NN * DD;
  const float* D2h = ws_nodes + (size_t)6 * NN * DD;
  const float* tauN = ws_nodes + (size_t)7 * NN * DD;

  float a = D1h[idx] - V1h[idx];
  float f = a * (cnt[i] * tauN[idx]) - fab[idx];  // f0 - fab
  red[n] = f * f;
  __syncthreads();
  for (int s = DD / 2; s > 0; s >>= 1) {
    if (n < s) red[n] += red[n + s];
    __syncthreads();
  }
  if (n == 0) s_norm = sqrtf(red[0]);
  __syncthreads();
  float fn = s_norm;
  float dtc = dt[i];
  float ho = V2h[idx] + f * dtc;
  float po = P3h[idx] + f * dtc + 0.5f * f * dtc * dtc;
  float dn = D2h[idx] + f / (fn + 1e-9f);
  out_h[idx] = h[idx] + lrelu(ho);
  out_p[idx] = p[idx] + lrelu(po);
  out_d[idx] = dv[idx] + lrelu(dn);
}

extern "C" void kernel_launch(void* const* d_in, const int* in_sizes, int n_in,
                              void* d_out, int out_size, void* d_ws, size_t ws_size,
                              hipStream_t stream) {
  const float* h   = (const float*)d_in[0];
  const float* e   = (const float*)d_in[1];
  const float* p   = (const float*)d_in[2];
  const float* dv  = (const float*)d_in[3];
  const float* dt  = (const float*)d_in[4];
  const int* smask = (const int*)d_in[5];
  const int* src   = (const int*)d_in[6];
  const int* dst   = (const int*)d_in[7];
  const float* W_V1=(const float*)d_in[8];  const float* b_V1=(const float*)d_in[9];
  const float* W_V2=(const float*)d_in[10]; const float* b_V2=(const float*)d_in[11];
  const float* W_E1=(const float*)d_in[12]; const float* b_E1=(const float*)d_in[13];
  const float* W_P1=(const float*)d_in[14]; const float* b_P1=(const float*)d_in[15];
  const float* W_P2=(const float*)d_in[16]; const float* b_P2=(const float*)d_in[17];
  const float* W_P3=(const float*)d_in[18]; const float* b_P3=(const float*)d_in[19];
  const float* W_D1=(const float*)d_in[20]; const float* b_D1=(const float*)d_in[21];
  const float* W_D2=(const float*)d_in[22]; const float* b_D2=(const float*)d_in[23];
  const float* W_V =(const float*)d_in[24]; const float* b_V =(const float*)d_in[25];
  const float* W_T =(const float*)d_in[26]; const float* b_T =(const float*)d_in[27];

  // Workspace: [0..6]=node GEMM outs, [7]=tau_node, [8]=fab, then cnt[NN], then f16 weights
  float* ws_nodes = (float*)d_ws;
  float* fab = ws_nodes + (size_t)8 * NN * DD;
  float* cnt = ws_nodes + (size_t)9 * NN * DD;
  _Float16* w16 = (_Float16*)(cnt + NN);  // node g*DD*DD, [7]=wE1, [8]=wV, [9..]=wT
  _Float16* wE1 = w16 + (size_t)7 * DD * DD;
  _Float16* wV  = w16 + (size_t)8 * DD * DD;

  const float* nodeW[7] = {W_V1, W_V2, W_P1, W_P2, W_P3, W_D1, W_D2};
  for (int g = 0; g < 7; ++g)
    cvt_w_kernel<<<(DD * DD + 255) / 256, 256, 0, stream>>>(nodeW[g], w16 + (size_t)g * DD * DD, DD * DD);
  cvt_w_kernel<<<(DD * DD + 255) / 256, 256, 0, stream>>>(W_E1, wE1, DD * DD);
  cvt_w_kernel<<<(DD * DD + 255) / 256, 256, 0, stream>>>(W_V, wV, DD * DD);
  cvt_w_kernel<<<(2 * DD * DD + 255) / 256, 256, 0, stream>>>(W_T, w16 + (size_t)9 * DD * DD, 2 * DD * DD);

  zero_kernel<<<(NN * DD + 255) / 256, 256, 0, stream>>>(fab, NN * DD);
  zero_kernel<<<(NN + 255) / 256, 256, 0, stream>>>(cnt, NN);
  dst_count_kernel<<<(NE + 255) / 256, 256, 0, stream>>>(dst, smask, cnt);

  dim3 ngrid(NN / 16, 1, 8);
  node_gemm_kernel<<<ngrid, 128, 0, stream>>>(h, p, dv, w16, b_V1, b_V2, b_P1, b_P2,
                                              b_P3, b_D1, b_D2, b_T, ws_nodes);

  float* out_h = (float*)d_out;
  float* out_e = out_h + (size_t)NN * DD;
  float* out_p = out_e + (size_t)NE * DD;
  float* out_d = out_p + (size_t)NN * DD;

  edge_kernel<<<NE / 16, 128, 0, stream>>>(h, e, dv, src, dst, smask, ws_nodes, fab,
                                           wE1, wV, b_E1, b_V, out_e);

  final_kernel<<<NN, 128, 0, stream>>>(h, p, dv, dt, ws_nodes, fab, cnt,
                                       out_h, out_p, out_d);
}
